// CausalTemporalAttnBlock_62783831933252
// MI455X (gfx1250) — matrix operand order repaired and checked by reading, not compiled, not run on hardware
//
#include <hip/hip_runtime.h>
#include <hip/hip_bf16.h>

// CausalTemporalAttnBlock for MI455X (gfx1250, wave32, WMMA + TDM).
// b=2, c=512, t=64, H=W=32. Fully fused per-(b,y,x) attention in 280KB LDS.

#define B_  2
#define C_  512
#define T_  64
#define H_  32
#define W_  32
#define EPS_ 1e-6f

typedef __bf16 bf16_t;
typedef __attribute__((ext_vector_type(16))) __bf16 v16bf;
typedef __attribute__((ext_vector_type(8)))  __bf16 v8bf;
typedef __attribute__((ext_vector_type(8)))  float  v8f;
typedef __attribute__((ext_vector_type(4)))  unsigned int u32x4;
typedef __attribute__((ext_vector_type(4)))  int i32x4;
typedef __attribute__((ext_vector_type(8)))  int i32x8;

// ---------------------------------------------------------------- WMMA helpers

__device__ __forceinline__ v8f wmma_bf16(v16bf a, v16bf b, v8f c) {
  // D = A(16x32 bf16) * B(32x16 bf16) + C(16x16 f32)
  return __builtin_amdgcn_wmma_f32_16x16x32_bf16(
      /*neg_a=*/false, a, /*neg_b=*/false, b,
      /*c_mod=*/(short)0, c, /*reuse_a=*/false, /*reuse_b=*/false);
}

// A fragment, 16-bit A 16x32 (ISA 7.12.2): lane -> row M=(lane&15),
// kb = (lane>>4)*8; halfs 0..7 = K[kb..kb+7], halfs 8..15 = K[kb+16..kb+23].
__device__ __forceinline__ v16bf load_a(const bf16_t* base, int ld, int m0,
                                        int k0, int lane) {
  const int m  = m0 + (lane & 15);
  const int kb = (lane >> 4) << 3;
  const bf16_t* p = base + m * ld + k0 + kb;
  v8bf lo = *(const v8bf*)(p);
  v8bf hi = *(const v8bf*)(p + 16);
  v16bf a;
#pragma unroll
  for (int i = 0; i < 8; ++i) { a[i] = lo[i]; a[i + 8] = hi[i]; }
  return a;
}

// B fragment, 16-bit B 32x16: lane -> col N=(lane&15), kb=(lane>>4)*8;
// halfs 0..7 = K[kb..kb+7], halfs 8..15 = K[kb+16..kb+23]. Requires K-major storage (row n holds K contiguous).
__device__ __forceinline__ v16bf load_b(const bf16_t* base, int ld, int n0,
                                        int k0, int lane) {
  const int n  = n0 + (lane & 15);
  const int kb = (lane >> 4) << 3;
  const bf16_t* p = base + n * ld + k0 + kb;
  v8bf lo = *(const v8bf*)(p);
  v8bf hi = *(const v8bf*)(p + 16);
  v16bf b;
#pragma unroll
  for (int i = 0; i < 8; ++i) { b[i] = lo[i]; b[i + 8] = hi[i]; }
  return b;
}

// D/C layout: element r of lane l -> M = r + 8*(l>>4), N = l&15.
__device__ __forceinline__ void store_d_row(bf16_t* base, int ld, int m0,
                                            int n0, int lane, v8f acc,
                                            float bias) {
  const int n  = n0 + (lane & 15);
  const int mb = m0 + ((lane >> 4) << 3);
#pragma unroll
  for (int r = 0; r < 8; ++r)
    base[(mb + r) * ld + n] = (bf16_t)(acc[r] + bias);
}

// Transposed store (for V): rows r are contiguous -> one ds_store_b128/lane.
__device__ __forceinline__ void store_d_col(bf16_t* base, int ld, int m0,
                                            int n0, int lane, v8f acc,
                                            float bias) {
  const int n  = n0 + (lane & 15);
  const int mb = m0 + ((lane >> 4) << 3);
  v8bf p;
#pragma unroll
  for (int r = 0; r < 8; ++r) p[r] = (bf16_t)(acc[r] + bias);
  *(v8bf*)(base + n * ld + mb) = p;
}

// ---------------------------------------------------------------- stats pass

__global__ __launch_bounds__(256) void stats_partial_kernel(
    const float* __restrict__ x, float* __restrict__ partial) {
  __shared__ float ssum[256];
  __shared__ float ssq[256];
  const int tid = threadIdx.x;
  const int blk = blockIdx.x;                       // 2048 = B_*1024
  const size_t base = (size_t)blk * 32768;          // elems per block
  const float4* p = (const float4*)(x + base);
  float s = 0.f, q = 0.f;
  for (int i = tid; i < 8192; i += 256) {
    float4 v = p[i];
    s += v.x + v.y + v.z + v.w;
    q += v.x * v.x + v.y * v.y + v.z * v.z + v.w * v.w;
  }
  ssum[tid] = s; ssq[tid] = q;
  __syncthreads();
  for (int st = 128; st > 0; st >>= 1) {
    if (tid < st) { ssum[tid] += ssum[tid + st]; ssq[tid] += ssq[tid + st]; }
    __syncthreads();
  }
  if (tid == 0) { partial[blk] = ssum[0]; partial[2048 + blk] = ssq[0]; }
}

__global__ __launch_bounds__(256) void stats_final_kernel(
    const float* __restrict__ partial, float* __restrict__ stats) {
  __shared__ float ssum[256];
  __shared__ float ssq[256];
  const int tid = threadIdx.x;
  const int b = blockIdx.x;
  float s = 0.f, q = 0.f;
  for (int i = tid; i < 1024; i += 256) {
    s += partial[b * 1024 + i];
    q += partial[2048 + b * 1024 + i];
  }
  ssum[tid] = s; ssq[tid] = q;
  __syncthreads();
  for (int st = 128; st > 0; st >>= 1) {
    if (tid < st) { ssum[tid] += ssum[tid + st]; ssq[tid] += ssq[tid + st]; }
    __syncthreads();
  }
  if (tid == 0) {
    const float n = (float)((size_t)C_ * T_ * H_ * W_);
    float mean = ssum[0] / n;
    float var  = ssq[0] / n - mean * mean;
    stats[b]     = mean;
    stats[2 + b] = rsqrtf(var + EPS_);
  }
}

// ---------------------------------------------------------------- weight cvt

__global__ __launch_bounds__(256) void cvt_weights_kernel(
    const float* __restrict__ wq, const float* __restrict__ wk,
    const float* __restrict__ wv, const float* __restrict__ wp,
    bf16_t* __restrict__ dst) {
  const int i = blockIdx.x * 256 + threadIdx.x;     // grid 1024 -> 262144
  dst[i]                = (bf16_t)wq[i];
  dst[262144 + i]       = (bf16_t)wk[i];
  dst[2 * 262144 + i]   = (bf16_t)wv[i];
  dst[3 * 262144 + i]   = (bf16_t)wp[i];
}

// ------------------------------------------------- normalize + transpose pass
// x[b,c,t,y,x] (f32) -> hT[(b,y,x)][t][c] (bf16), GroupNorm applied.

__global__ __launch_bounds__(256) void norm_transpose_kernel(
    const float* __restrict__ x, const float* __restrict__ gamma,
    const float* __restrict__ beta, const float* __restrict__ stats,
    bf16_t* __restrict__ hT) {
  __shared__ bf16_t tile[C_ * 33];                  // 512 x 32, pad to 33
  const int bid = blockIdx.x;                       // B_*T_*H_ = 4096
  const int b   = bid / (T_ * H_);
  const int rem = bid % (T_ * H_);
  const int t   = rem / H_;
  const int y   = rem % H_;
  const float mean = stats[b];
  const float rstd = stats[2 + b];
  const int tid = threadIdx.x, w = tid >> 5, lane = tid & 31;

  // coalesced read: lanes span x (innermost dim)
  for (int c = w; c < C_; c += 8) {
    float v = x[((((size_t)b * C_ + c) * T_ + t) * H_ + y) * W_ + lane];
    v = (v - mean) * rstd * gamma[c] + beta[c];
    tile[c * 33 + lane] = (bf16_t)v;
  }
  __syncthreads();

  // coalesced write: each wave emits one x-row of 512 contiguous bf16
  for (int xw = w; xw < W_; xw += 8) {
    bf16_t* dst = hT + ((((size_t)b * H_ + y) * W_ + xw) * T_ + t) * (size_t)C_;
    v16bf o;
#pragma unroll
    for (int j = 0; j < 16; ++j) o[j] = tile[(lane * 16 + j) * 33 + xw];
    *(v16bf*)(dst + lane * 16) = o;
  }
}

// ---------------------------------------------------- fused attention kernel
// One workgroup (8 wave32) per (b,y,x) location. Everything stays in LDS.

#define SMEM_BYTES (4 * (T_ * C_ * 2) + T_ * T_ * 4 + T_ * T_ * 2)  // 286720

__global__ __launch_bounds__(256, 1) void attn_kernel(
    const float* __restrict__ x,
    const bf16_t* __restrict__ wq, const bf16_t* __restrict__ wk,
    const bf16_t* __restrict__ wv, const bf16_t* __restrict__ wp,
    const float* __restrict__ bq, const float* __restrict__ bk,
    const float* __restrict__ bv, const float* __restrict__ bp,
    const bf16_t* __restrict__ hT, float* __restrict__ out) {
  extern __shared__ char smem[];
  bf16_t* h_lds  = (bf16_t*)smem;            // [t][c] 64x512 (later reused as O)
  bf16_t* q_lds  = h_lds + T_ * C_;          // [t][c]
  bf16_t* k_lds  = q_lds + T_ * C_;          // [s][c]
  bf16_t* vT_lds = k_lds + T_ * C_;          // [c][s] (transposed V)
  float*  s_lds  = (float*)(vT_lds + C_ * T_);   // [t][s] scores 64x64 f32
  bf16_t* a_lds  = (bf16_t*)(s_lds + T_ * T_);   // [t][s] attn bf16

  const int loc  = blockIdx.x;               // b*1024 + y*32 + x
  const int b    = loc >> 10;
  const int y    = (loc >> 5) & 31;
  const int xw   = loc & 31;
  const int tid  = threadIdx.x;
  const int wid  = tid >> 5;
  const int lane = tid & 31;

  // ---- TDM: DMA the 64KB contiguous h tile into LDS with one instruction ----
  // D# (ISA 8.3/8.4): 2-D tensor, data_size=8B, tensor_dim0=tile_dim0=8192,
  // stride=8192, tile_dim1=1; groups 2/3 zero (<=2D). Issued by wave 0 only
  // (EXEC-independent, tracked by TENSORcnt); barrier publishes to all waves.
  if (wid == 0) {
    const unsigned lds_base = (unsigned)(size_t)h_lds;  // low 32b = LDS byte addr
    const unsigned long long ga =
        (unsigned long long)(const void*)(hT + (size_t)loc * (T_ * C_));
    u32x4 g0;
    g0[0] = 1u;                                        // count=1 user descriptor
    g0[1] = lds_base;                                  // lds_addr
    g0[2] = (unsigned)(ga & 0xffffffffu);              // global_addr[31:0]
    g0[3] = (unsigned)((ga >> 32) & 0x01ffffffu)       // global_addr[56:32]
            | (2u << 30);                              // type=2 ("image")
    i32x8 g1;
    g1[0] = (3 << 16);            // workgroup_mask=0, data_size=3 (8B)
    g1[1] = (int)(8192u << 16);   // tensor_dim0[15:0]=8192 in bits[31:16]
    g1[2] = (int)(1u << 16);      // tensor_dim0 hi=0; tensor_dim1=1
    g1[3] = (int)(8192u << 16);   // tensor_dim1 hi=0; tile_dim0=8192
    g1[4] = 1;                    // tile_dim1=1, tile_dim2=0
    g1[5] = 8192;                 // tensor_dim0_stride[31:0]
    g1[6] = (int)(8192u << 16);   // stride0 hi=0; tensor_dim1_stride lo=8192
    g1[7] = 0;                    // tensor_dim1_stride hi
    i32x4 z4 = {};
#if __clang_major__ >= 23
    i32x8 z8 = {};
    __builtin_amdgcn_tensor_load_to_lds(g0, g1, z4, z4, z8, 0);
#else
    __builtin_amdgcn_tensor_load_to_lds(g0, g1, z4, z4, 0);
#endif
    __builtin_amdgcn_s_wait_tensorcnt((short)0);
  }
  __syncthreads();

  // ---- Q, K, V projections: (64x512) = h(64x512) @ W^T, W is [o][c] K-major ----
  const bf16_t* Wm[3] = {wq, wk, wv};
  const float*  Bm[3] = {bq, bk, bv};
#pragma unroll
  for (int m = 0; m < 3; ++m) {
    const bf16_t* Wt = Wm[m];
    const float*  bi = Bm[m];
#pragma unroll 1
    for (int i = 0; i < 16; ++i) {           // 128 tiles / 8 waves
      const int flat = i * 8 + wid;
      const int tt = flat >> 5, oo = flat & 31;
      v8f acc = {};
#pragma unroll 4
      for (int k0 = 0; k0 < C_; k0 += 32) {
        __builtin_prefetch((const void*)(Wt + (size_t)(oo * 16) * C_ + k0 + 128), 0, 1);
        v16bf a  = load_a(h_lds, C_, tt * 16, k0, lane);
        v16bf bb = load_b(Wt,   C_, oo * 16, k0, lane);
        acc = wmma_bf16(a, bb, acc);
      }
      const float bo = bi[oo * 16 + (lane & 15)];
      if (m == 0)      store_d_row(q_lds,  C_, tt * 16, oo * 16, lane, acc, bo);
      else if (m == 1) store_d_row(k_lds,  C_, tt * 16, oo * 16, lane, acc, bo);
      else             store_d_col(vT_lds, T_, tt * 16, oo * 16, lane, acc, bo);
    }
  }
  __syncthreads();

  // ---- scores = scale * Q K^T : 16 tiles (4x4) over K=512 ----
  const float scale = 0.044194173824159216f;  // 512^-0.5
#pragma unroll
  for (int i = 0; i < 2; ++i) {
    const int flat = i * 8 + wid;
    const int tt = flat >> 2, ss = flat & 3;
    v8f acc = {};
#pragma unroll 4
    for (int k0 = 0; k0 < C_; k0 += 32) {
      v16bf a  = load_a(q_lds, C_, tt * 16, k0, lane);
      v16bf bb = load_b(k_lds, C_, ss * 16, k0, lane);   // col s, K=c contiguous
      acc = wmma_bf16(a, bb, acc);
    }
    const int n  = ss * 16 + (lane & 15);
    const int mb = tt * 16 + ((lane >> 4) << 3);
#pragma unroll
    for (int r = 0; r < 8; ++r) s_lds[(mb + r) * T_ + n] = acc[r] * scale;
  }
  __syncthreads();

  // ---- causal softmax, one wave per 8 rows, wave32 shuffle reductions ----
#pragma unroll 1
  for (int r = 0; r < 8; ++r) {
    const int t = wid * 8 + r;
    const int s0 = lane * 2, s1 = lane * 2 + 1;
    float v0 = (s0 <= t) ? s_lds[t * T_ + s0] : -1e30f;
    float v1 = (s1 <= t) ? s_lds[t * T_ + s1] : -1e30f;
    float mx = fmaxf(v0, v1);
#pragma unroll
    for (int off = 16; off > 0; off >>= 1) mx = fmaxf(mx, __shfl_xor(mx, off, 32));
    float e0 = (s0 <= t) ? __expf(v0 - mx) : 0.f;
    float e1 = (s1 <= t) ? __expf(v1 - mx) : 0.f;
    float sm = e0 + e1;
#pragma unroll
    for (int off = 16; off > 0; off >>= 1) sm += __shfl_xor(sm, off, 32);
    const float inv = 1.f / sm;
    a_lds[t * T_ + s0] = (bf16_t)(e0 * inv);
    a_lds[t * T_ + s1] = (bf16_t)(e1 * inv);
  }
  __syncthreads();

  // ---- O = attn(64x64) @ V(64x512): B from vT_lds (K=s contiguous). Reuse h_lds. ----
#pragma unroll 1
  for (int i = 0; i < 16; ++i) {
    const int flat = i * 8 + wid;
    const int tt = flat >> 5, oo = flat & 31;
    v8f acc = {};
#pragma unroll
    for (int k0 = 0; k0 < T_; k0 += 32) {
      v16bf a  = load_a(a_lds,  T_, tt * 16, k0, lane);
      v16bf bb = load_b(vT_lds, T_, oo * 16, k0, lane);
      acc = wmma_bf16(a, bb, acc);
    }
    store_d_row(h_lds, C_, tt * 16, oo * 16, lane, acc, 0.f);
  }
  __syncthreads();

  // ---- Y = O @ Wp^T + bp ; out = x + Y (residual) ----
#pragma unroll 1
  for (int i = 0; i < 16; ++i) {
    const int flat = i * 8 + wid;
    const int tt = flat >> 5, oo = flat & 31;
    v8f acc = {};
#pragma unroll 4
    for (int k0 = 0; k0 < C_; k0 += 32) {
      __builtin_prefetch((const void*)(wp + (size_t)(oo * 16) * C_ + k0 + 128), 0, 1);
      v16bf a  = load_a(h_lds, C_, tt * 16, k0, lane);
      v16bf bb = load_b(wp,   C_, oo * 16, k0, lane);
      acc = wmma_bf16(a, bb, acc);
    }
    const int o  = oo * 16 + (lane & 15);
    const int mb = tt * 16 + ((lane >> 4) << 3);
    const float bo = bp[o];
#pragma unroll
    for (int r = 0; r < 8; ++r) {
      const int t = mb + r;
      const size_t idx =
          ((((size_t)b * C_ + o) * T_ + t) * H_ + y) * W_ + xw;
      out[idx] = x[idx] + acc[r] + bo;   // scattered 4B; merged in 192MB L2
    }
  }
}

// ---------------------------------------------------------------- launcher

extern "C" void kernel_launch(void* const* d_in, const int* in_sizes, int n_in,
                              void* d_out, int out_size, void* d_ws,
                              size_t ws_size, hipStream_t stream) {
  const float* x     = (const float*)d_in[0];
  const float* gamma = (const float*)d_in[1];
  const float* beta  = (const float*)d_in[2];
  const float* wq    = (const float*)d_in[3];
  const float* bq    = (const float*)d_in[4];
  const float* wk    = (const float*)d_in[5];
  const float* bk    = (const float*)d_in[6];
  const float* wv    = (const float*)d_in[7];
  const float* bv    = (const float*)d_in[8];
  const float* wp    = (const float*)d_in[9];
  const float* bp    = (const float*)d_in[10];
  float* out = (float*)d_out;

  char* ws = (char*)d_ws;
  float*  partial = (float*)ws;                       // 4096 f32
  float*  stats   = partial + 4096;                   // mean[2], rstd[2]
  bf16_t* wbf     = (bf16_t*)(ws + 65536);            // 4 x 512x512 bf16
  bf16_t* hT      = (bf16_t*)(ws + 65536 + (size_t)4 * 262144 * 2);

  stats_partial_kernel<<<2048, 256, 0, stream>>>(x, partial);
  stats_final_kernel<<<2, 256, 0, stream>>>(partial, stats);
  cvt_weights_kernel<<<1024, 256, 0, stream>>>(wq, wk, wv, wp, wbf);
  norm_transpose_kernel<<<4096, 256, 0, stream>>>(x, gamma, beta, stats, hT);

  static int attr_set = 0;  // idempotent host-side attribute (not a stream op)
  if (!attr_set) {
    hipFuncSetAttribute((const void*)attn_kernel,
                        hipFuncAttributeMaxDynamicSharedMemorySize, SMEM_BYTES);
    attr_set = 1;
  }
  attn_kernel<<<B_ * H_ * W_, 256, SMEM_BYTES, stream>>>(
      x, wbf, wbf + 262144, wbf + 2 * 262144, wbf + 3 * 262144,
      bq, bk, bv, bp, hT, out);
}
